// MultiHeadSelfAttention_283467842633
// MI455X (gfx1250) — matrix operand order repaired and checked by reading, not compiled, not run on hardware
//
#include <hip/hip_runtime.h>

// ---------------------------------------------------------------------------
// MI455X (gfx1250) multi-head self-attention, bf16 WMMA pipeline with
// double-buffered GLOBAL_LOAD_ASYNC_TO_LDS staging.
// B=8, S=1024, D=768, H=12, Hd=64.  All matmuls via v_wmma_f32_16x16x32_bf16.
// ---------------------------------------------------------------------------

typedef __bf16 bf16;
typedef __bf16 v16bf __attribute__((ext_vector_type(16)));
typedef __bf16 v8bf  __attribute__((ext_vector_type(8)));
typedef float  v8f   __attribute__((ext_vector_type(8)));

constexpr int BATCH   = 8;
constexpr int SEQ     = 1024;
constexpr int DMODEL  = 768;
constexpr int NHEAD   = 12;
constexpr int HDIM    = 64;
constexpr int ROWS    = BATCH * SEQ;        // 8192

// ---------------------------------------------------------------------------
// Async global->LDS copy (16 bytes / lane), tracked by ASYNCcnt.
// vdst carries the LDS byte address (low 32 bits of the generic pointer are
// the LDS offset per the gfx1250 aperture mapping).
// ---------------------------------------------------------------------------
__device__ __forceinline__ void async_copy_b128(void* lds_dst, const void* gsrc) {
    unsigned lds_off = (unsigned)(unsigned long long)lds_dst;
    asm volatile("global_load_async_to_lds_b128 %0, %1, off"
                 :: "v"(lds_off), "v"((unsigned long long)gsrc) : "memory");
}
__device__ __forceinline__ void wait_async0() {
    asm volatile("s_wait_asynccnt 0" ::: "memory");
}

// ---------------------------------------------------------------------------
// Fragment helpers (wave32, 16x16x32 bf16 WMMA VGPR layouts per CDNA5 ISA)
// ---------------------------------------------------------------------------

// A-matrix 16x32: lane m = l%16; lanes 0-15 hold K 0-7 & 16-23, lanes 16-31
// hold K 8-15 & 24-31.  prow points at (row m, k=0).  Two 16B loads.
__device__ __forceinline__ v16bf load_afrag(const bf16* prow, int lane) {
    int kh = ((lane >> 4) & 1) * 8;
    v8bf lo = *(const v8bf*)(prow + kh);
    v8bf hi = *(const v8bf*)(prow + 16 + kh);
    return __builtin_shufflevector(lo, hi, 0,1,2,3,4,5,6,7,8,9,10,11,12,13,14,15);
}

// B-matrix 32x16: lane n = l%16; lanes 0-15 hold K 0-7 & 16-23, lanes 16-31 hold K 8-15 & 24-31.
// prow points at (row n, k=0) of a row-major [N][K] (i.e. transposed) tile.
__device__ __forceinline__ v16bf load_bfrag(const bf16* prow, int lane) {
    int kg = (lane >> 4) * 8;
    v8bf lo = *(const v8bf*)(prow + kg);
    v8bf hi = *(const v8bf*)(prow + kg + 16);
    return __builtin_shufflevector(lo, hi, 0,1,2,3,4,5,6,7,8,9,10,11,12,13,14,15);
}

__device__ __forceinline__ v8f wmma_bf16(v16bf a, v16bf b, v8f c) {
    return __builtin_amdgcn_wmma_f32_16x16x32_bf16(
        /*neg_a=*/false, a, /*neg_b=*/false, b,
        /*c_mod=*/(short)0, c, /*reuse_a=*/false, /*reuse_b=*/false);
}

// ---------------------------------------------------------------------------
// Kernel 1a: fp32 -> bf16 conversion (row-major copy; n multiple of 4)
// ---------------------------------------------------------------------------
__global__ void mha_cvt_bf16(const float* __restrict__ in, bf16* __restrict__ out, int n) {
    int i = (blockIdx.x * blockDim.x + threadIdx.x) * 4;
    if (i + 3 < n) {
        float4 f = *(const float4*)(in + i);
        out[i + 0] = (bf16)f.x;
        out[i + 1] = (bf16)f.y;
        out[i + 2] = (bf16)f.z;
        out[i + 3] = (bf16)f.w;
    }
}

// ---------------------------------------------------------------------------
// Kernel 1b: fp32 [K][N] -> bf16 [N][K] transposed weights (one-off, so the
// GEMM's B-tile staging becomes pure contiguous 16B copies every k-step).
// ---------------------------------------------------------------------------
__global__ void mha_cvt_transpose(const float* __restrict__ in, bf16* __restrict__ out,
                                  int K, int N) {
    int id = blockIdx.x * blockDim.x + threadIdx.x;
    if (id < K * N) {
        int k = id / N;
        int n = id - k * N;
        out[(size_t)n * K + k] = (bf16)in[id];
    }
}

// ---------------------------------------------------------------------------
// Kernel 2/4: tiled WMMA GEMM  C[M,N] = A[M,K] * BT[N,K]^T + bias
//   block tile 256x64, K-step 32, 256 threads = 8 waves, wave owns 32xN rows
//   (8 WMMAs / k-step / wave).  Double-buffered async LDS staging.
//   mode 0: QKV epilogue -> scatter bf16 into q/k/v [B,H,S,Hd], q *= 0.125
//   mode 1: proj epilogue -> fp32 store to outf
// ---------------------------------------------------------------------------
__global__ __launch_bounds__(256)
void mha_gemm_wmma(const bf16* __restrict__ A, const bf16* __restrict__ BT,
                   int M, int N, int K, const float* __restrict__ bias, int mode,
                   bf16* __restrict__ qb, bf16* __restrict__ kb, bf16* __restrict__ vb,
                   float* __restrict__ outf) {
    __shared__ bf16 Xs[2][256][40];   // A tile, row-major, 80B rows (16B aligned)
    __shared__ bf16 Wt[2][64][40];    // B tile, [n][k]

    const int t    = threadIdx.x;
    const int lane = t & 31;
    const int wave = t >> 5;
    const int row0 = blockIdx.y * 256;
    const int n0   = blockIdx.x * 64;

    auto stage = [&](int buf, int k0) {
        // A tile: 256x32 bf16 = 1024 x 16B, 4 per thread (async DMA to LDS)
        #pragma unroll
        for (int i = 0; i < 4; ++i) {
            int lin = i * 256 + t;          // 0..1023
            int r   = lin >> 2;             // 0..255
            int c8  = (lin & 3) * 8;        // 0,8,16,24
            async_copy_b128(&Xs[buf][r][c8],
                            A + (size_t)(row0 + r) * K + k0 + c8);
        }
        // B tile (pre-transposed): 64x32 bf16 = 256 x 16B, 1 per thread
        {
            int n  = t >> 2;                // 0..63
            int k8 = (t & 3) * 8;           // 0,8,16,24
            async_copy_b128(&Wt[buf][n][k8],
                            BT + (size_t)(n0 + n) * K + k0 + k8);
        }
    };

    v8f acc[8];
    #pragma unroll
    for (int i = 0; i < 8; ++i) acc[i] = v8f{};

    const int nk = K / 32;
    stage(0, 0);
    for (int kt = 0; kt < nk; ++kt) {
        const int cur = kt & 1;
        wait_async0();          // this wave's DMA for buf 'cur' has landed
        __syncthreads();        // everyone's has; prior readers of buf^1 done
        if (kt + 1 < nk) stage(cur ^ 1, (kt + 1) * 32);   // overlap next DMA

        v16bf a0 = load_afrag(&Xs[cur][wave * 32 + (lane & 15)][0], lane);
        v16bf a1 = load_afrag(&Xs[cur][wave * 32 + 16 + (lane & 15)][0], lane);
        #pragma unroll
        for (int nb = 0; nb < 4; ++nb) {
            v16bf bfr = load_bfrag(&Wt[cur][nb * 16 + (lane & 15)][0], lane);
            acc[nb]     = wmma_bf16(a0, bfr, acc[nb]);
            acc[4 + nb] = wmma_bf16(a1, bfr, acc[4 + nb]);
        }
        __syncthreads();        // all reads of 'cur' done before it is re-staged
    }

    // epilogue: C layout -> row = v + 8*(lane>=16), col = lane%16
    const int hi8   = (lane >> 4) * 8;
    const int nlane = lane & 15;
    #pragma unroll
    for (int f = 0; f < 8; ++f) {
        const int nb   = f & 3;
        const int half = f >> 2;
        #pragma unroll
        for (int v = 0; v < 8; ++v) {
            int mg = row0 + wave * 32 + half * 16 + hi8 + v;
            int ng = n0 + nb * 16 + nlane;
            float val = acc[f][v] + bias[ng];
            if (mode == 0) {
                int which = ng / DMODEL;
                int d     = ng % DMODEL;
                int h     = d >> 6;
                int hd    = d & 63;
                int bi    = mg >> 10;       // /SEQ
                int s     = mg & 1023;      // %SEQ
                size_t idx = (((size_t)bi * NHEAD + h) * SEQ + s) * HDIM + hd;
                if (which == 0)      qb[idx] = (bf16)(val * 0.125f);   // Hd^-0.5
                else if (which == 1) kb[idx] = (bf16)val;
                else                 vb[idx] = (bf16)val;
            } else {
                outf[(size_t)mg * N + ng] = val;
            }
        }
    }
}

// ---------------------------------------------------------------------------
// Kernel 3: flash attention.  grid = (qtiles=8, B*H=96), 256 threads.
//   Each wave: 16 query rows x full Hd, online softmax over 1024 keys in
//   32-key chunks.  K chunks async-DMA'd to LDS (double-buffered); V chunks
//   transposed into LDS manually.
// ---------------------------------------------------------------------------
__global__ __launch_bounds__(256)
void mha_flash_attn(const bf16* __restrict__ qb, const bf16* __restrict__ kb,
                    const bf16* __restrict__ vb, bf16* __restrict__ ob) {
    __shared__ bf16 Ks[2][32][72];     // [key][hd], 144B rows (16B aligned)
    __shared__ bf16 Vt[2][64][40];     // [hd][key] transposed
    __shared__ bf16 Pl[8][16][40];     // per-wave P staging (C -> A relayout)

    const int t    = threadIdx.x;
    const int lane = t & 31;
    const int wave = t >> 5;
    const int bh   = blockIdx.y;               // 0..95
    const int b    = bh / NHEAD;
    const int h    = bh % NHEAD;
    const int q0   = blockIdx.x * 128 + wave * 16;   // this wave's first q row

    const bf16* qbh = qb + (size_t)bh * SEQ * HDIM;
    const bf16* kbh = kb + (size_t)bh * SEQ * HDIM;
    const bf16* vbh = vb + (size_t)bh * SEQ * HDIM;

    auto stage = [&](int buf, int kc) {
        int key = t >> 3;                  // 0..31
        int hd8 = (t & 7) * 8;             // 0..56
        async_copy_b128(&Ks[buf][key][hd8],
                        kbh + (size_t)(kc * 32 + key) * HDIM + hd8);
        v8bf vv = *(const v8bf*)(vbh + (size_t)(kc * 32 + key) * HDIM + hd8);
        #pragma unroll
        for (int j = 0; j < 8; ++j) Vt[buf][hd8 + j][key] = vv[j];
    };

    // Q fragments: 16 rows x 64 (Hd) = two A frags (k 0-31, 32-63)
    const bf16* qrow = qbh + (size_t)(q0 + (lane & 15)) * HDIM;
    v16bf qa0 = load_afrag(qrow, lane);
    v16bf qa1 = load_afrag(qrow + 32, lane);

    v8f o[4] = {v8f{}, v8f{}, v8f{}, v8f{}};   // 16 x 64 output accum
    float m_i[8], l_i[8];
    #pragma unroll
    for (int v = 0; v < 8; ++v) { m_i[v] = -1e30f; l_i[v] = 0.0f; }

    const int colA = lane & 15;
    const int hi8  = (lane >> 4) * 8;
    constexpr int NCHUNK = SEQ / 32;

    stage(0, 0);
    for (int kc = 0; kc < NCHUNK; ++kc) {
        const int cur = kc & 1;
        wait_async0();
        __syncthreads();
        if (kc + 1 < NCHUNK) stage(cur ^ 1, kc + 1);   // overlap next K/V DMA

        // ---- S = Q * K^T for 16q x 32k (4 WMMAs) ----
        v8f s0 = v8f{}, s1 = v8f{};
        {
            const bf16* r0 = &Ks[cur][(lane & 15)][0];        // keys 0-15
            const bf16* r1 = &Ks[cur][16 + (lane & 15)][0];   // keys 16-31
            s0 = wmma_bf16(qa0, load_bfrag(r0,      lane), s0);
            s0 = wmma_bf16(qa1, load_bfrag(r0 + 32, lane), s0);
            s1 = wmma_bf16(qa0, load_bfrag(r1,      lane), s1);
            s1 = wmma_bf16(qa1, load_bfrag(r1 + 32, lane), s1);
        }

        // ---- online softmax (row stats live in the lane group owning the row)
        float alpha[8];
        #pragma unroll
        for (int v = 0; v < 8; ++v) {
            float pm = fmaxf(s0[v], s1[v]);
            #pragma unroll
            for (int off = 1; off < 16; off <<= 1)
                pm = fmaxf(pm, __shfl_xor(pm, off, 32));
            float mn = fmaxf(m_i[v], pm);
            alpha[v] = __expf(m_i[v] - mn);
            float p0 = __expf(s0[v] - mn);
            float p1 = __expf(s1[v] - mn);
            s0[v] = p0; s1[v] = p1;
            float rs = p0 + p1;
            #pragma unroll
            for (int off = 1; off < 16; off <<= 1)
                rs += __shfl_xor(rs, off, 32);
            l_i[v] = l_i[v] * alpha[v] + rs;
            m_i[v] = mn;
        }
        #pragma unroll
        for (int j = 0; j < 4; ++j)
            #pragma unroll
            for (int v = 0; v < 8; ++v)
                o[j][v] = o[j][v] * alpha[v];

        // ---- relayout P: C-fragment -> bf16 A-fragment via per-wave LDS ----
        #pragma unroll
        for (int v = 0; v < 8; ++v) {
            Pl[wave][hi8 + v][colA]      = (bf16)s0[v];
            Pl[wave][hi8 + v][16 + colA] = (bf16)s1[v];
        }
        asm volatile("s_wait_dscnt 0" ::: "memory");   // intra-wave LDS RAW
        v16bf pA = load_afrag(&Pl[wave][lane & 15][0], lane);

        // ---- O += P * V (4 WMMAs, one per 16-wide Hd block) ----
        #pragma unroll
        for (int j = 0; j < 4; ++j) {
            v16bf vfr = load_bfrag(&Vt[cur][j * 16 + (lane & 15)][0], lane);
            o[j] = wmma_bf16(pA, vfr, o[j]);
        }
    }

    // ---- normalize and write to [B,S,D] bf16 for the output projection ----
    #pragma unroll
    for (int j = 0; j < 4; ++j) {
        #pragma unroll
        for (int v = 0; v < 8; ++v) {
            int r = q0 + hi8 + v;
            size_t idx = ((size_t)b * SEQ + r) * DMODEL + h * HDIM + j * 16 + colA;
            ob[idx] = (bf16)(o[j][v] * (1.0f / l_i[v]));
        }
    }
}

// ---------------------------------------------------------------------------
// Launch
// ---------------------------------------------------------------------------
extern "C" void kernel_launch(void* const* d_in, const int* in_sizes, int n_in,
                              void* d_out, int out_size, void* d_ws, size_t ws_size,
                              hipStream_t stream) {
    const float* x      = (const float*)d_in[0];
    const float* w_qkv  = (const float*)d_in[1];
    const float* b_qkv  = (const float*)d_in[2];
    const float* w_proj = (const float*)d_in[3];
    const float* b_proj = (const float*)d_in[4];
    float* out          = (float*)d_out;

    // workspace carve-up (bf16 buffers, ~66 MB total)
    constexpr size_t N_XB   = (size_t)ROWS * DMODEL;          // 8192*768
    constexpr size_t N_WQKV = (size_t)DMODEL * 3 * DMODEL;    // 768*2304
    constexpr size_t N_WPRJ = (size_t)DMODEL * DMODEL;        // 768*768
    constexpr size_t N_HEAD = (size_t)BATCH * NHEAD * SEQ * HDIM;

    char* p = (char*)d_ws;
    bf16* xb      = (bf16*)p; p += N_XB   * sizeof(bf16);
    bf16* wqkvT   = (bf16*)p; p += N_WQKV * sizeof(bf16);   // [3D][D] transposed
    bf16* wprojT  = (bf16*)p; p += N_WPRJ * sizeof(bf16);   // [D][D] transposed
    bf16* qb      = (bf16*)p; p += N_HEAD * sizeof(bf16);
    bf16* kb      = (bf16*)p; p += N_HEAD * sizeof(bf16);
    bf16* vb      = (bf16*)p; p += N_HEAD * sizeof(bf16);
    bf16* ob      = (bf16*)p; p += N_XB   * sizeof(bf16);

    {   // x: straight bf16 copy
        int blocks = (int)((N_XB / 4 + 255) / 256);
        mha_cvt_bf16<<<blocks, 256, 0, stream>>>(x, xb, (int)N_XB);
    }
    {   // weights: convert + transpose once so GEMM staging is contiguous
        int blocks = (int)((N_WQKV + 255) / 256);
        mha_cvt_transpose<<<blocks, 256, 0, stream>>>(w_qkv, wqkvT, DMODEL, 3 * DMODEL);
        blocks = (int)((N_WPRJ + 255) / 256);
        mha_cvt_transpose<<<blocks, 256, 0, stream>>>(w_proj, wprojT, DMODEL, DMODEL);
    }

    // QKV projection: [8192,768] x [768,2304]
    mha_gemm_wmma<<<dim3(3 * DMODEL / 64, ROWS / 256), 256, 0, stream>>>(
        xb, wqkvT, ROWS, 3 * DMODEL, DMODEL, b_qkv, /*mode=*/0,
        qb, kb, vb, nullptr);

    // flash attention: 8 q-tiles x 96 (b,h)
    mha_flash_attn<<<dim3(SEQ / 128, BATCH * NHEAD), 256, 0, stream>>>(qb, kb, vb, ob);

    // output projection: [8192,768] x [768,768] -> fp32 + bias
    mha_gemm_wmma<<<dim3(DMODEL / 64, ROWS / 256), 256, 0, stream>>>(
        ob, wprojT, ROWS, DMODEL, DMODEL, b_proj, /*mode=*/1,
        nullptr, nullptr, nullptr, out);
}
